// SSConv2d_55542517072567
// MI455X (gfx1250) — matrix operand order repaired and checked by reading, not compiled, not run on hardware
//
#include <hip/hip_runtime.h>

typedef unsigned int   u32;
typedef unsigned short u16;
typedef unsigned long long u64;
typedef __attribute__((ext_vector_type(4)))  u32    u32x4;
typedef __attribute__((ext_vector_type(8)))  u32    u32x8;
typedef __attribute__((ext_vector_type(16))) __bf16 v16bf;
typedef __attribute__((ext_vector_type(8)))  float  v8f;

namespace {
constexpr int  kB    = 32,  kGin = 64, kH = 28, kW = 28;
constexpr int  kCin  = 512, kCout = 512;
constexpr int  kHp   = 30,  kWp = 30;             // padded spatial
constexpr int  kPix  = kH * kW;                   // 784
constexpr int  kMtot = kB * kPix;                 // 25088 (GEMM M)
constexpr int  kKC   = kCin / 32;                 // 16 ci chunks of 32 (WMMA K)
constexpr int  kNT   = kCout / 16;                // 32 co tiles of 16 (WMMA N)
constexpr long kDnElems = (long)kB * kHp * kWp * kCin;  // 14,745,600 bf16
constexpr int  kOutVals = kB * kGin * kH * kW;          // 1,605,632 per output tensor
}

__device__ __forceinline__ u16 f2bf(float f) {
  // round-to-nearest-even f32 -> bf16 bits
  u32 u = __builtin_bit_cast(u32, f);
  u32 r = u + 0x7FFFu + ((u >> 16) & 1u);
  return (u16)(r >> 16);
}

__device__ __forceinline__ v16bf load_frag32B(const u16* p0, const u16* p1) {
  u32x4 lo = *reinterpret_cast<const u32x4*>(p0);
  u32x4 hi = *reinterpret_cast<const u32x4*>(p1);
  u32x8 c;
  c[0] = lo[0]; c[1] = lo[1]; c[2] = lo[2]; c[3] = lo[3];
  c[4] = hi[0]; c[5] = hi[1]; c[6] = hi[2]; c[7] = hi[3];
  return __builtin_bit_cast(v16bf, c);
}

// ---------------- kernel 1: zero padded dense buffer ----------------
__global__ void zero_dn_kernel(u32x4* __restrict__ p, long n4) {
  long i = (long)blockIdx.x * blockDim.x + threadIdx.x;
  if (i < n4) {
    u32x4 z = {0u, 0u, 0u, 0u};
    p[i] = z;
  }
}

// ---------------- kernel 2: scatter sparse blocks -> dense NHWC bf16 ----------------
__global__ void scatter_dense_kernel(const float* __restrict__ inp,
                                     const int* __restrict__ idx,
                                     u16* __restrict__ dn) {
  int i = blockIdx.x * blockDim.x + threadIdx.x;
  if (i >= kB * kGin * kH * kW) return;
  int x = i % kW; int t = i / kW;
  int y = t % kH; t /= kH;
  int g = t % kGin; int b = t / kGin;
  u16 bv = f2bf(inp[i]);
  int id  = idx[i] & 7;
  u32 w[4] = {0u, 0u, 0u, 0u};
  w[id >> 1] = (u32)bv << ((id & 1) * 16);
  u32x4 st = {w[0], w[1], w[2], w[3]};
  long off = (((long)b * kHp + (y + 1)) * kWp + (x + 1)) * kCin + g * 8;
  *reinterpret_cast<u32x4*>(dn + off) = st;  // 16B aligned (g*8 bf16)
}

// ---------------- kernel 3: pack weights into WMMA B-fragment lane order ----------------
// Wp[kpos][cic][coTile][lane][16] bf16 : lane<16 -> K=cic*32+0..15, lane>=16 -> K=+16..31,
// N = coTile*16 + (lane&15). One lane row = 32 contiguous bytes.
__global__ void pack_weights_kernel(const float* __restrict__ wgt, u16* __restrict__ wp) {
  int tid = blockIdx.x * blockDim.x + threadIdx.x;
  if (tid >= 9 * kKC * kNT * 32) return;
  int lane   = tid & 31;
  int coTile = (tid >> 5) % kNT;
  int cic    = (tid / (32 * kNT)) % kKC;
  int kpos   = tid / (32 * kNT * kKC);
  int kh = kpos / 3, kw = kpos % 3;
  int co    = coTile * 16 + (lane & 15);
  int kbase = (lane >> 4) * 16;
  u32 w[8];
#pragma unroll
  for (int j = 0; j < 8; ++j) {
    int ci0 = cic * 32 + kbase + 2 * j;
    float f0 = wgt[(((long)co * kCin + ci0) * 3 + kh) * 3 + kw];
    float f1 = wgt[(((long)co * kCin + ci0 + 1) * 3 + kh) * 3 + kw];
    w[j] = (u32)f2bf(f0) | ((u32)f2bf(f1) << 16);
  }
  u32x4 lo = {w[0], w[1], w[2], w[3]};
  u32x4 hi = {w[4], w[5], w[6], w[7]};
  u32x4* dst = reinterpret_cast<u32x4*>(wp + (long)tid * 16);
  dst[0] = lo;
  dst[1] = hi;
}

// ---------------- kernel 4: implicit-GEMM conv + fused bias + block max/argmax ----------
// grid = (196, 2): 128-pixel x 256-channel WG tiles. 8 waves = 2 (M) x 4 (N);
// each wave computes a 64x64 tile (4 m-tiles x 4 n-tiles, 128 acc VGPRs).
__launch_bounds__(256, 1)
__global__ void conv_wmma_kernel(const u16* __restrict__ dn, const u16* __restrict__ wp,
                                 const float* __restrict__ bias, float* __restrict__ out) {
  const int lane   = threadIdx.x & 31;
  const int wave   = threadIdx.x >> 5;
  const int laneHi = lane >> 4;           // A-layout: lanes 16-31 hold K +16..31 (contiguous)
  const int laneN  = lane & 15;
  const int wm     = wave & 1;            // 2 M groups
  const int wn     = wave >> 1;           // 4 N groups
  const int pbase  = blockIdx.x * 128 + wm * 64;
  const int coBase = blockIdx.y * 256 + wn * 64;

  int pb[4], py[4], px[4], rowOff[4];
#pragma unroll
  for (int t = 0; t < 4; ++t) {
    int p = pbase + t * 16 + laneN;       // A row M = lane%16
    pb[t] = p / kPix;
    int q = p % kPix;
    py[t] = q / kW;
    px[t] = q % kW;
    rowOff[t] = ((pb[t] * kHp + py[t]) * kWp + px[t]) * kCin + laneHi * 16;
  }

  v8f acc[4][4];
#pragma unroll
  for (int t = 0; t < 4; ++t)
#pragma unroll
    for (int n = 0; n < 4; ++n)
      acc[t][n] = v8f{0.f, 0.f, 0.f, 0.f, 0.f, 0.f, 0.f, 0.f};

#pragma unroll 1
  for (int kpos = 0; kpos < 9; ++kpos) {
    const int kh = kpos / 3, kw = kpos % 3;
    const int kOff = (kh * kWp + kw) * kCin;
    // prefetch the next tap's activation rows into cache (global_prefetch_b8)
    if (kpos < 8) {
      const int kh2 = (kpos + 1) / 3, kw2 = (kpos + 1) % 3;
      const int kOff2 = (kh2 * kWp + kw2) * kCin;
#pragma unroll
      for (int t = 0; t < 4; ++t)
        __builtin_prefetch(dn + rowOff[t] + kOff2, 0, 3);
    }
    const u16* wk = wp + (((long)kpos * kKC * kNT) + (coBase >> 4)) * 512 + lane * 16;

#pragma unroll 1
    for (int cic = 0; cic < kKC; ++cic) {
      const u16* wcic = wk + (long)cic * kNT * 512;
      // prefetch next K-step's weight tiles
      __builtin_prefetch(wcic + kNT * 512, 0, 3);
      v16bf afrag[4];
#pragma unroll
      for (int t = 0; t < 4; ++t) {
        const u16* pa = dn + rowOff[t] + kOff + cic * 32;
        afrag[t] = load_frag32B(pa, pa + 8);    // K run base..base+15 (contiguous)
      }
#pragma unroll
      for (int n = 0; n < 4; ++n) {
        const u16* pw = wcic + n * 512;         // 512 elems per 32x16 tile
        v16bf bfrag = load_frag32B(pw, pw + 8);
#pragma unroll
        for (int t = 0; t < 4; ++t)
          acc[t][n] = __builtin_amdgcn_wmma_f32_16x16x32_bf16(
              false, afrag[t], false, bfrag, (short)0, acc[t][n], false, false);
      }
    }
  }

  // Fused epilogue: bias add, then max over 8-channel blocks via fmax butterflies;
  // argmax recovered with one ballot (lowest set lane == first max index, matching
  // jnp.argmax tie-breaking).
#pragma unroll
  for (int n = 0; n < 4; ++n) {
    float bv = bias[coBase + n * 16 + laneN];
#pragma unroll
    for (int t = 0; t < 4; ++t) {
#pragma unroll
      for (int r = 0; r < 8; ++r) {
        float v = acc[t][n][r] + bv;
        float vmax = v;
#pragma unroll
        for (int m = 1; m < 8; m <<= 1)
          vmax = fmaxf(vmax, __shfl_xor(vmax, m, 32));
        u64 mask = __ballot(v == vmax);
        if ((lane & 7) == 0) {
          u32 grp = (u32)(mask >> (lane & 24)) & 0xFFu;
          int id  = __builtin_ctz(grp | 0x100u);   // lowest max lane in this 8-group
          int M = laneHi ? (r + 8) : r;            // C layout: lanes 16-31 hold M=8..15
          int p = pbase + t * 16 + M;
          int b = p / kPix, q = p % kPix;
          int y = q / kW, x = q % kW;
          int g = (coBase >> 3) + n * 2 + ((lane >> 3) & 1);
          int o = (((b * kGin) + g) * kH + y) * kW + x;
          out[o] = vmax;
          out[kOutVals + o] = (float)id;
        }
      }
    }
  }
}

extern "C" void kernel_launch(void* const* d_in, const int* in_sizes, int n_in,
                              void* d_out, int out_size, void* d_ws, size_t ws_size,
                              hipStream_t stream) {
  const float* inp  = (const float*)d_in[0];
  const int*   idx  = (const int*)d_in[1];
  const float* wgt  = (const float*)d_in[2];
  const float* bias = (const float*)d_in[3];
  float* out = (float*)d_out;

  u16* dn = (u16*)d_ws;            // 29,491,200 B
  u16* wq = dn + kDnElems;         // + 4,718,592 B  (total ~34.2 MB of d_ws)

  long n4 = kDnElems / 8;
  zero_dn_kernel<<<(unsigned)((n4 + 255) / 256), 256, 0, stream>>>((u32x4*)dn, n4);

  int nsc = kB * kGin * kH * kW;
  scatter_dense_kernel<<<(nsc + 255) / 256, 256, 0, stream>>>(inp, idx, dn);

  int npk = 9 * kKC * kNT * 32;
  pack_weights_kernel<<<(npk + 255) / 256, 256, 0, stream>>>(wgt, wq);

  dim3 grid(kMtot / 128, kCout / 256);   // (196, 2)
  conv_wmma_kernel<<<grid, 256, 0, stream>>>(dn, wq, bias, out);
}
